// MultiHeadAttention_7103875907963
// MI455X (gfx1250) — matrix operand order repaired and checked by reading, not compiled, not run on hardware
//
#include <hip/hip_runtime.h>
#include <hip/hip_bf16.h>
#include <math.h>

// MHA forward for B=2, T=2048, C=1024, H=16, D=64 on gfx1250 (wave32, WMMA).
typedef __attribute__((ext_vector_type(16))) _Float16 v16h;
typedef __attribute__((ext_vector_type(8)))  _Float16 v8h;
typedef __attribute__((ext_vector_type(8)))  float    v8f;

#define NTOK 4096   // B*T
#define CDIM 1024
#define TSEQ 2048
#define HEADS 16
#define HDIM 64

// ---------------------------------------------------------------------------
// WMMA helpers.  V_WMMA_F32_16X16X32_F16 fragment layouts used here (A/B contiguous k, not ISA table order):
//   A (16x32 f16): lane L holds row M=L&15; elem j -> K = j + 16*(L>>4)
//   B (32x16 f16): lane L holds col N=L&15; elem j -> K = j + 16*(L>>4)
//   C/D (16x16 f32): lane L holds col N=L&15; elem r -> M = r + 8*(L>>4)
// ---------------------------------------------------------------------------
__device__ inline v8f wmma_f16(v16h a, v16h b, v8f c) {
  return __builtin_amdgcn_wmma_f32_16x16x32_f16(false, a, false, b, (short)0, c,
                                                false, false);
}

// A fragment from row-major [.., ld] f16 matrix, tile origin (m0, k0).
// Two contiguous 16B loads per lane (global_load_b128 / ds_load_b128).
__device__ inline v16h load_frag_a(const _Float16* __restrict__ p, int ld,
                                   int m0, int k0) {
  int lane = threadIdx.x & 31;
  const _Float16* q = p + (m0 + (lane & 15)) * ld + k0 + 16 * (lane >> 4);
  v8h lo = *(const v8h*)q;         // elems 0..7  : K = kbase + 0..7
  v8h hi = *(const v8h*)(q + 8);   // elems 8..15 : K = kbase + 8..15
  v16h a;
#pragma unroll
  for (int i = 0; i < 8; ++i) { a[i] = lo[i]; a[i + 8] = hi[i]; }
  return a;
}

// B fragment where B(K,N) = mem[N*ld + K]  (i.e. "row-major [N][K]" storage,
// which is exactly x@W^T with W row-major, and K[t][d] / Vt[d][t]).
// One contiguous 32B load per lane.
__device__ inline v16h load_frag_b(const _Float16* __restrict__ p, int ld,
                                   int k0, int n0) {
  int lane = threadIdx.x & 31;
  const _Float16* q = p + (n0 + (lane & 15)) * ld + k0 + 16 * (lane >> 4);
  return *(const v16h*)q;          // elems j=0..15 : K = kbase + j
}

// ---------------------------------------------------------------------------
// f32 -> f16 cast, 8 elements/thread, b128 in / b128 out.  n % 8 == 0.
// ---------------------------------------------------------------------------
__global__ void cast_f32_f16_kernel(const float* __restrict__ src,
                                    _Float16* __restrict__ dst, int n) {
  int stride = gridDim.x * blockDim.x;
  for (int idx = blockIdx.x * blockDim.x + threadIdx.x; idx * 8 < n; idx += stride) {
    const float4* s = (const float4*)src + idx * 2;
    float4 a = s[0], b = s[1];
    v8h o;
    o[0] = (_Float16)a.x; o[1] = (_Float16)a.y; o[2] = (_Float16)a.z; o[3] = (_Float16)a.w;
    o[4] = (_Float16)b.x; o[5] = (_Float16)b.y; o[6] = (_Float16)b.z; o[7] = (_Float16)b.w;
    *((v8h*)dst + idx) = o;
  }
}

// ---------------------------------------------------------------------------
// Fused Q/K/V projection:  [4096,1024]f16 x W^T + bias.
// One wave per 16x64 output STRIP (4 accumulators).  All 10 b128 loads of a
// k-step are issued into DISTINCT registers before the 4 WMMAs so the waits
// stay partial and loads for step k+1 overlap the matrix ops of step k.
// A 64-wide strip is exactly one head:  Q,K stored per-head [B,H,T,D];
// V stored transposed [B,H,D,T] (so the PV WMMA B-fragments are contiguous).
// The V transpose is free: D-fragment rows are consecutive tokens -> one
// packed 16B store per lane per sub-tile.
// ---------------------------------------------------------------------------
__global__ void __launch_bounds__(128) qkv_proj_kernel(
    const _Float16* __restrict__ xh,
    const _Float16* __restrict__ wq, const _Float16* __restrict__ wk,
    const _Float16* __restrict__ wv,
    const float* __restrict__ bq, const float* __restrict__ bk,
    const float* __restrict__ bv,
    _Float16* __restrict__ qh, _Float16* __restrict__ kh,
    _Float16* __restrict__ vt) {
  int lane = threadIdx.x & 31;
  int wid  = blockIdx.x * 4 + (threadIdx.x >> 5);
  int mat  = wid / (256 * 16);      // 0=Q, 1=K, 2=V
  int rem  = wid % (256 * 16);
  int m0   = (rem >> 4) << 4;       // 256 token tiles
  int n0   = (rem & 15) << 6;       // 16 strips of 64 channels (one head)
  const _Float16* w   = (mat == 0) ? wq : (mat == 1) ? wk : wv;
  const float*    bia = (mat == 0) ? bq : (mat == 1) ? bk : bv;

  v8f acc0 = {}, acc1 = {}, acc2 = {}, acc3 = {};

#pragma unroll 4
  for (int k0 = 0; k0 < CDIM; k0 += 32) {
    v16h a  = load_frag_a(xh, CDIM, m0, k0);
    v16h b0 = load_frag_b(w, CDIM, k0, n0);
    v16h b1 = load_frag_b(w, CDIM, k0, n0 + 16);
    v16h b2 = load_frag_b(w, CDIM, k0, n0 + 32);
    v16h b3 = load_frag_b(w, CDIM, k0, n0 + 48);
    acc0 = wmma_f16(a, b0, acc0);
    acc1 = wmma_f16(a, b1, acc1);
    acc2 = wmma_f16(a, b2, acc2);
    acc3 = wmma_f16(a, b3, acc3);
  }

  v8f acc[4] = {acc0, acc1, acc2, acc3};
  int hi = lane >> 4;
  int h  = n0 >> 6;                 // head index for this strip

  if (mat < 2) {
    _Float16* outp = (mat == 0) ? qh : kh;
#pragma unroll
    for (int nt = 0; nt < 4; ++nt) {
      int d = nt * 16 + (lane & 15);
      float bsc = bia[n0 + d];
#pragma unroll
      for (int r = 0; r < 8; ++r) {
        int token = m0 + r + 8 * hi;
        int bb = token >> 11, t = token & (TSEQ - 1);
        outp[(((bb * HEADS + h) * TSEQ) + t) * HDIM + d] =
            (_Float16)(acc[nt][r] + bsc);
      }
    }
  } else {
    int bb = m0 >> 11;
    int t0 = (m0 & (TSEQ - 1)) + 8 * hi;
#pragma unroll
    for (int nt = 0; nt < 4; ++nt) {
      int d = nt * 16 + (lane & 15);
      float bsc = bia[n0 + d];
      v8h pk;
#pragma unroll
      for (int r = 0; r < 8; ++r) pk[r] = (_Float16)(acc[nt][r] + bsc);
      *(v8h*)(vt + ((bb * HEADS + h) * HDIM + d) * TSEQ + t0) = pk;
    }
  }
}

// ---------------------------------------------------------------------------
// Flash attention, causal.  One wave per (b,h, 16-query tile); 32-key blocks.
// S = Q K^T via 4 WMMAs/block, online softmax with 4-step __shfl_xor
// butterflies (a softmax row lives across 16 lanes of the wave in D-layout),
// P transposed D-layout -> A-layout through a 1KB/wave LDS tile,
// O += P V via 4 WMMAs/block against transposed V.  The 4 V fragments are
// fetched BEFORE the softmax VALU section so the ~100 VALU ops of the online
// update hide their L2 latency.
// ---------------------------------------------------------------------------
__global__ void __launch_bounds__(128) attn_kernel(
    const _Float16* __restrict__ qh, const _Float16* __restrict__ kh,
    const _Float16* __restrict__ vt, _Float16* __restrict__ yh) {
  __shared__ __align__(32) _Float16 plds[4][16][32];
  int lane = threadIdx.x & 31;
  int wv   = threadIdx.x >> 5;
  int wid  = blockIdx.x * 4 + wv;
  int bh   = wid >> 7;              // 0..31 = b*16 + h
  int m0   = (wid & 127) << 4;      // query tile start within [0,T)
  int b    = bh >> 4, h = bh & 15;
  const _Float16* qp = qh + bh * TSEQ * HDIM;
  const _Float16* kp = kh + bh * TSEQ * HDIM;
  const _Float16* vp = vt + bh * HDIM * TSEQ;
  const int hi = lane >> 4;
  const float scale = 0.125f;       // 1/sqrt(64)

  // Q tile stays resident: two A fragments cover d = 0..63.
  v16h aq0 = load_frag_a(qp, HDIM, m0, 0);
  v16h aq1 = load_frag_a(qp, HDIM, m0, 32);

  float m_i[8], l_i[8];
  v8f acc[4];
  v8f zero = {};
#pragma unroll
  for (int r = 0; r < 8; ++r) { m_i[r] = -INFINITY; l_i[r] = 0.0f; }
#pragma unroll
  for (int nt = 0; nt < 4; ++nt) acc[nt] = zero;

  for (int j0 = 0; j0 <= m0 + 15; j0 += 32) {   // causal: keys <= max query
    // K fragments for the 32-key block (distinct registers, batched loads).
    v16h k00 = load_frag_b(kp, HDIM, 0,  j0);
    v16h k01 = load_frag_b(kp, HDIM, 32, j0);
    v16h k10 = load_frag_b(kp, HDIM, 0,  j0 + 16);
    v16h k11 = load_frag_b(kp, HDIM, 32, j0 + 16);
    // V fragments: independent of S, issued now and consumed after softmax.
    v16h bv0 = load_frag_b(vp, TSEQ, j0, 0);
    v16h bv1 = load_frag_b(vp, TSEQ, j0, 16);
    v16h bv2 = load_frag_b(vp, TSEQ, j0, 32);
    v16h bv3 = load_frag_b(vp, TSEQ, j0, 48);

    v8f s0 = zero, s1 = zero;
    s0 = wmma_f16(aq0, k00, s0);
    s0 = wmma_f16(aq1, k01, s0);
    s1 = wmma_f16(aq0, k10, s1);
    s1 = wmma_f16(aq1, k11, s1);

    int key0 = j0 + (lane & 15);
#pragma unroll
    for (int r = 0; r < 8; ++r) {
      int query = m0 + r + 8 * hi;
      float v0 = (key0      <= query) ? s0[r] * scale : -INFINITY;
      float v1 = (key0 + 16 <= query) ? s1[r] * scale : -INFINITY;
      float mx = fmaxf(v0, v1);
#pragma unroll
      for (int off = 1; off < 16; off <<= 1) mx = fmaxf(mx, __shfl_xor(mx, off, 32));
      float mnew = fmaxf(m_i[r], mx);
      float corr = __expf(m_i[r] - mnew);
      float p0 = __expf(v0 - mnew);
      float p1 = __expf(v1 - mnew);
      float sum = p0 + p1;
#pragma unroll
      for (int off = 1; off < 16; off <<= 1) sum += __shfl_xor(sum, off, 32);
      l_i[r] = l_i[r] * corr + sum;
      m_i[r] = mnew;
#pragma unroll
      for (int nt = 0; nt < 4; ++nt) acc[nt][r] *= corr;
      int row = r + 8 * hi;
      plds[wv][row][lane & 15]        = (_Float16)p0;
      plds[wv][row][16 + (lane & 15)] = (_Float16)p1;
    }

    // Re-read P in A-fragment layout (same-wave DS ops are in-order).
    v16h ap = load_frag_a(&plds[wv][0][0], 32, 0, 0);
    acc[0] = wmma_f16(ap, bv0, acc[0]);
    acc[1] = wmma_f16(ap, bv1, acc[1]);
    acc[2] = wmma_f16(ap, bv2, acc[2]);
    acc[3] = wmma_f16(ap, bv3, acc[3]);
  }

#pragma unroll
  for (int r = 0; r < 8; ++r) {
    float inv = 1.0f / l_i[r];
    int row = b * TSEQ + m0 + r + 8 * hi;
#pragma unroll
    for (int nt = 0; nt < 4; ++nt)
      yh[row * CDIM + h * HDIM + nt * 16 + (lane & 15)] =
          (_Float16)(acc[nt][r] * inv);
  }
}

// ---------------------------------------------------------------------------
// Output projection: y[4096,1024]f16 x Wo^T + bo -> out f32.
// Same 16x64-strip-per-wave structure as the QKV projection.
// ---------------------------------------------------------------------------
__global__ void __launch_bounds__(128) out_proj_kernel(
    const _Float16* __restrict__ yh, const _Float16* __restrict__ wo,
    const float* __restrict__ bo, float* __restrict__ out) {
  int lane = threadIdx.x & 31;
  int wid  = blockIdx.x * 4 + (threadIdx.x >> 5);
  int m0   = (wid >> 4) << 4;       // 256 token tiles
  int n0   = (wid & 15) << 6;       // 16 strips of 64 channels

  v8f acc0 = {}, acc1 = {}, acc2 = {}, acc3 = {};

#pragma unroll 4
  for (int k0 = 0; k0 < CDIM; k0 += 32) {
    v16h a  = load_frag_a(yh, CDIM, m0, k0);
    v16h b0 = load_frag_b(wo, CDIM, k0, n0);
    v16h b1 = load_frag_b(wo, CDIM, k0, n0 + 16);
    v16h b2 = load_frag_b(wo, CDIM, k0, n0 + 32);
    v16h b3 = load_frag_b(wo, CDIM, k0, n0 + 48);
    acc0 = wmma_f16(a, b0, acc0);
    acc1 = wmma_f16(a, b1, acc1);
    acc2 = wmma_f16(a, b2, acc2);
    acc3 = wmma_f16(a, b3, acc3);
  }

  v8f acc[4] = {acc0, acc1, acc2, acc3};
  int hi = lane >> 4;
#pragma unroll
  for (int nt = 0; nt < 4; ++nt) {
    int co = n0 + nt * 16 + (lane & 15);
    float bsc = bo[co];
#pragma unroll
    for (int r = 0; r < 8; ++r)
      out[(m0 + r + 8 * hi) * CDIM + co] = acc[nt][r] + bsc;
  }
}

// ---------------------------------------------------------------------------
extern "C" void kernel_launch(void* const* d_in, const int* in_sizes, int n_in,
                              void* d_out, int out_size, void* d_ws, size_t ws_size,
                              hipStream_t stream) {
  (void)in_sizes; (void)n_in; (void)out_size; (void)ws_size;
  const float* x  = (const float*)d_in[0];
  // d_in[1] = attn_mask (causal tril) -> applied analytically, not read.
  const float* Wq = (const float*)d_in[2];
  const float* bq = (const float*)d_in[3];
  const float* Wk = (const float*)d_in[4];
  const float* bk = (const float*)d_in[5];
  const float* Wv = (const float*)d_in[6];
  const float* bv = (const float*)d_in[7];
  const float* Wo = (const float*)d_in[8];
  const float* bo = (const float*)d_in[9];
  float* out = (float*)d_out;

  // Workspace layout (f16 elements): 48 MB total, fits L2 (192 MB).
  _Float16* ws  = (_Float16*)d_ws;
  _Float16* xh  = ws;                 // [4096,1024]
  _Float16* wqh = ws + 4194304;       // [1024,1024]
  _Float16* wkh = ws + 5242880;
  _Float16* wvh = ws + 6291456;
  _Float16* woh = ws + 7340032;
  _Float16* qh  = ws + 8388608;       // [B,H,T,D]
  _Float16* kh  = ws + 12582912;      // [B,H,T,D]
  _Float16* vt  = ws + 16777216;      // [B,H,D,T] (transposed V)
  _Float16* yh  = ws + 20971520;      // [4096,1024] attention output

  cast_f32_f16_kernel<<<2048, 256, 0, stream>>>(x,  xh,  NTOK * CDIM);
  cast_f32_f16_kernel<<<512,  256, 0, stream>>>(Wq, wqh, CDIM * CDIM);
  cast_f32_f16_kernel<<<512,  256, 0, stream>>>(Wk, wkh, CDIM * CDIM);
  cast_f32_f16_kernel<<<512,  256, 0, stream>>>(Wv, wvh, CDIM * CDIM);
  cast_f32_f16_kernel<<<512,  256, 0, stream>>>(Wo, woh, CDIM * CDIM);

  // 256 Mtiles * 16 Nstrips * 3 matrices = 12288 waves, 4 waves/block.
  qkv_proj_kernel<<<3072, 128, 0, stream>>>(xh, wqh, wkh, wvh, bq, bk, bv,
                                            qh, kh, vt);
  // 32 (b,h) * 128 query tiles = 4096 waves.
  attn_kernel<<<1024, 128, 0, stream>>>(qh, kh, vt, yh);
  // 256 Mtiles * 16 Nstrips = 4096 waves.
  out_proj_kernel<<<1024, 128, 0, stream>>>(yh, woh, bo, out);
}
